// Block_120259084502
// MI455X (gfx1250) — matrix operand order repaired and checked by reading, not compiled, not run on hardware
//
#include <hip/hip_runtime.h>
#include <hip/hip_bf16.h>

// ---------------------------------------------------------------------------
// Transformer block for MI455X (gfx1250, wave32, WMMA + Tensor Data Mover).
// All GEMMs run on v_wmma_f32_16x16x32_bf16 (fp32 accum); GEMM/attention tiles
// are staged into LDS by TDM tensor_load_to_lds (TENSORcnt-tracked, double
// buffered) when the builtin is available, with a cooperative-load fallback.
// Shapes fixed by the reference: B=4, T=2048, E=1024, H=16, D=64, F=4096.
// ---------------------------------------------------------------------------

typedef __attribute__((ext_vector_type(16))) __bf16 v16bf;
typedef __attribute__((ext_vector_type(8)))  __bf16 v8bf;
typedef __attribute__((ext_vector_type(8)))  float  v8f;
typedef __attribute__((ext_vector_type(4)))  unsigned int v4u;
typedef __attribute__((ext_vector_type(8)))  int v8i;
typedef __attribute__((ext_vector_type(4)))  int v4i;

union FragA { v16bf v; v8bf h[2]; };

#define WMMA_BF16(a, b, c) \
  __builtin_amdgcn_wmma_f32_16x16x32_bf16(false, (a), false, (b), (short)0, (c), false, false)

#if defined(__AMDGCN__) && __has_builtin(__builtin_amdgcn_tensor_load_to_lds) && \
    __has_builtin(__builtin_amdgcn_s_wait_tensorcnt)
#define HAVE_TDM 1
#else
#define HAVE_TDM 0
#endif

__device__ inline unsigned lds_addr_of(const void* p) {
  // generic pointers to LDS carry the LDS byte address in the low 32 bits
  return (unsigned)(size_t)p;
}

#if HAVE_TDM
// Issue one TDM 2D tile load: tile_d0 x tile_d1 bf16 elements from a row-major
// tensor with row stride `stride_elems`, packed contiguously into LDS at
// lds_off. D# bit layout per CDNA5 ISA ch.8 (group0 128b, group1 256b).
__device__ inline void tdm_load_2d(const void* gsrc, unsigned lds_off,
                                   unsigned tile_d0, unsigned tile_d1,
                                   unsigned stride_elems) {
  const unsigned long long ga = (unsigned long long)(size_t)gsrc;
  const unsigned tdim0 = stride_elems;   // full row length (OOB bound)
  const unsigned tdim1 = 1u << 30;       // plenty of rows; tiles stay in range
  v4u g0;
  g0[0] = 1u;                                                   // count=1
  g0[1] = lds_off;                                              // LDS byte addr
  g0[2] = (unsigned)ga;                                         // global[31:0]
  g0[3] = (unsigned)((ga >> 32) & 0x1ffffffull) | (2u << 30);   // [56:32]+type=2
  v8i g1;
  g1[0] = (int)(1u << 16);                                      // data_size=2B
  g1[1] = (int)((tdim0 & 0xffffu) << 16);                       // tdim0[15:0]
  g1[2] = (int)(((tdim0 >> 16) & 0xffffu) | ((tdim1 & 0xffffu) << 16));
  g1[3] = (int)(((tdim1 >> 16) & 0xffffu) | ((tile_d0 & 0xffffu) << 16));
  g1[4] = (int)(tile_d1 & 0xffffu);                             // tile_d1, d2=0
  g1[5] = (int)stride_elems;                                    // d0 stride lo32
  g1[6] = 0;                                                    // stride hi, d1 stride
  g1[7] = 0;
  v4i z4 = {};
#if __clang_major__ >= 23
  v8i z8 = {};
  __builtin_amdgcn_tensor_load_to_lds(g0, g1, z4, z4, z8, 0);
#else
  __builtin_amdgcn_tensor_load_to_lds(g0, g1, z4, z4, 0);
#endif
}
#endif  // HAVE_TDM

// ---------------------------------------------------------------------------
// Weight packing: fp32 -> bf16, transposed to [N][K] so GEMM B-tiles are
// contiguous row reads.
// ---------------------------------------------------------------------------

// Wq/Wk/Wv: [H, E, D] fp32 -> [N=H*D][K=E] bf16  (out[n*E + k] = W[h][k][d])
__global__ void pack_qkv_kernel(const float* __restrict__ W, __bf16* __restrict__ out,
                                int E, int D) {
  int idx = blockIdx.x * 256 + threadIdx.x;
  if (idx >= E * E) return;
  int n  = idx / E;
  int kk = idx - n * E;
  int h  = n / D;
  int d  = n - h * D;
  out[idx] = (__bf16)W[((size_t)h * E + kk) * (size_t)D + d];
}

// Generic [K][N] fp32 -> [N][K] bf16
__global__ void pack_t_kernel(const float* __restrict__ W, __bf16* __restrict__ out,
                              int K, int N) {
  int idx = blockIdx.x * 256 + threadIdx.x;
  if (idx >= K * N) return;
  int n  = idx / K;
  int kk = idx - n * K;
  out[idx] = (__bf16)W[(size_t)kk * N + n];
}

// ---------------------------------------------------------------------------
// LayerNorm: one workgroup (256 thr = 8 waves) per row, bf16 output.
// ---------------------------------------------------------------------------
__global__ __launch_bounds__(256) void ln_kernel(const float* __restrict__ x,
                                                 const float* __restrict__ g,
                                                 const float* __restrict__ be,
                                                 __bf16* __restrict__ out, int E) {
  const int row = blockIdx.x;
  const float* xr = x + (size_t)row * E;
  float s = 0.f, sq = 0.f;
  for (int i = threadIdx.x; i < E; i += 256) {
    float v = xr[i];
    s += v; sq += v * v;
  }
#pragma unroll
  for (int off = 1; off < 32; off <<= 1) {
    s  += __shfl_xor(s,  off, 32);
    sq += __shfl_xor(sq, off, 32);
  }
  __shared__ float ss[8], sqs[8];
  const int wid = threadIdx.x >> 5;
  if ((threadIdx.x & 31) == 0) { ss[wid] = s; sqs[wid] = sq; }
  __syncthreads();
  float ts = 0.f, tq = 0.f;
#pragma unroll
  for (int i = 0; i < 8; ++i) { ts += ss[i]; tq += sqs[i]; }
  const float mean = ts / (float)E;
  const float var  = tq / (float)E - mean * mean;
  const float rsq  = rsqrtf(var + 1e-5f);
  __bf16* orow = out + (size_t)row * E;
  for (int i = threadIdx.x; i < E; i += 256)
    orow[i] = (__bf16)((xr[i] - mean) * rsq * g[i] + be[i]);
}

// ---------------------------------------------------------------------------
// WMMA GEMM: C[M,N] = act(A[M,K] @ B[K,N] + bias + res)
//   A: bf16 row-major [M][K];  Bt: bf16 pre-transposed [N][K].
// 64x64 workgroup tile, 128 threads = 4 waves; wave w owns rows w*16..+15 and
// four 16x16 accumulators. K stepped by 32 through double-buffered LDS slabs
// staged by TDM (wave 0 issues, TENSORcnt 2-deep pipelining).
// ---------------------------------------------------------------------------
template <bool BIAS, bool RES, bool RELU, bool OBF, bool OF32>
__global__ __launch_bounds__(128) void gemm_bf16_wmma(
    const __bf16* __restrict__ A, const __bf16* __restrict__ Bt,
    const float* __restrict__ bias, const float* __restrict__ res,
    __bf16* __restrict__ outb, float* __restrict__ outf,
    int M, int N, int K) {
  __shared__ __bf16 Al[2][64 * 32];
  __shared__ __bf16 Bl[2][64 * 32];

  const int tid  = threadIdx.x;
  const int wave = tid >> 5;
  const int lane = tid & 31;
  const int half = lane >> 4;
  const int l15  = lane & 15;
  const int m0 = blockIdx.y * 64;
  const int n0 = blockIdx.x * 64;

  const __bf16* aTile = A  + (size_t)m0 * K;
  const __bf16* bTile = Bt + (size_t)n0 * K;
  const int lr = tid >> 1;   // fallback loader row
  const int lh = tid & 1;
  (void)lr; (void)lh; (void)M;

  v8f acc[4] = {};

  // Stage the [64 x 32] K-slab starting at column kt into LDS buffer `buf`.
  auto load_tiles = [&](int buf, int kt) {
#if HAVE_TDM
    if (wave == 0) {
      tdm_load_2d(aTile + kt, lds_addr_of(&Al[buf][0]), 32u, 64u, (unsigned)K);
      tdm_load_2d(bTile + kt, lds_addr_of(&Bl[buf][0]), 32u, 64u, (unsigned)K);
    }
#else
    const __bf16* as = aTile + (size_t)lr * K + kt + lh * 16;
    const __bf16* bs = bTile + (size_t)lr * K + kt + lh * 16;
    __bf16* ad = &Al[buf][lr * 32 + lh * 16];
    __bf16* bd = &Bl[buf][lr * 32 + lh * 16];
    *(uint4*)(ad)     = *(const uint4*)(as);
    *(uint4*)(ad + 8) = *(const uint4*)(as + 8);
    *(uint4*)(bd)     = *(const uint4*)(bs);
    *(uint4*)(bd + 8) = *(const uint4*)(bs + 8);
    if (kt + 32 < K) {
      __builtin_prefetch(as + 32, 0, 1);
      __builtin_prefetch(bs + 32, 0, 1);
    }
#endif
  };

  load_tiles(0, 0);
  int cur = 0;
  for (int kt = 0; kt < K; kt += 32) {
    if (kt + 32 < K) {
      load_tiles(cur ^ 1, kt + 32);       // prefetch next slab into other buffer
#if HAVE_TDM
      if (wave == 0) __builtin_amdgcn_s_wait_tensorcnt(2);  // current slab done
#endif
    } else {
#if HAVE_TDM
      if (wave == 0) __builtin_amdgcn_s_wait_tensorcnt(0);
#endif
    }
    __syncthreads();

    // A fragment 16x32 (wave32 layout: half0 K{0..7,16..23}, half1 +8)
    FragA af;
    const __bf16* ab = &Al[cur][(wave * 16 + l15) * 32];
    af.h[0] = *(const v8bf*)(ab + half * 8);
    af.h[1] = *(const v8bf*)(ab + 16 + half * 8);
    // Batch all four B fragments, then the four WMMAs (one dscnt wait total)
    FragA bfr[4];
#pragma unroll
    for (int nt = 0; nt < 4; ++nt) {
      const __bf16* bb = &Bl[cur][(nt * 16 + l15) * 32 + half * 8];
      bfr[nt].h[0] = *(const v8bf*)(bb);
      bfr[nt].h[1] = *(const v8bf*)(bb + 16);
    }
#pragma unroll
    for (int nt = 0; nt < 4; ++nt)
      acc[nt] = WMMA_BF16(af.v, bfr[nt].v, acc[nt]);

    __syncthreads();   // everyone done with buf[cur] before it is re-filled
    cur ^= 1;
  }

  // Epilogue. C layout: VGPR r holds row r + 8*half, lane&15 = column.
#pragma unroll
  for (int nt = 0; nt < 4; ++nt) {
    const int col = n0 + nt * 16 + l15;
    const float bc = BIAS ? bias[col] : 0.f;
#pragma unroll
    for (int r = 0; r < 8; ++r) {
      const int row = m0 + wave * 16 + r + 8 * half;
      float v = acc[nt][r] + bc;
      if (RES)  v += res[(size_t)row * N + col];
      if (RELU) v  = fmaxf(v, 0.f);
      if (OF32) outf[(size_t)row * N + col] = v;
      if (OBF)  outb[(size_t)row * N + col] = (__bf16)v;
    }
  }
}

// ---------------------------------------------------------------------------
// Flash attention (non-causal), D=64. Grid: (T/64, B*H), 128 threads.
// Each wave owns a 16-row Q strip; loop over S in 64-wide tiles with online
// softmax. K tile (row-major, TDM-staged) -> QK^T B-frags contiguous; V tile
// stored transposed [d][s] -> PV B-frags contiguous. P converts C-layout ->
// A-layout through per-wave LDS.
// ---------------------------------------------------------------------------
__global__ __launch_bounds__(128) void attn_flash_wmma(
    const __bf16* __restrict__ Q, const __bf16* __restrict__ Kg,
    const __bf16* __restrict__ Vg, __bf16* __restrict__ O,
    int T, int E, int H) {
  __shared__ __bf16 Kt[64 * 64];
  __shared__ __bf16 Vt[64 * 64];       // transposed: Vt[d][s]
  __shared__ __bf16 Pl[4 * 16 * 64];   // per-wave P staging

  const int tid  = threadIdx.x;
  const int wave = tid >> 5;
  const int lane = tid & 31;
  const int half = lane >> 4;
  const int l15  = lane & 15;
  const int t0 = blockIdx.x * 64;
  const int bh = blockIdx.y;
  const int b = bh / H, h = bh - b * H;
  const size_t rowBase = (size_t)b * T;
  const int c0 = h * 64;

  // Q fragments for this wave's 16 rows, D=64 -> two 16x32 A-frags
  FragA qf[2];
  {
    const __bf16* qr = Q + (rowBase + t0 + wave * 16 + l15) * (size_t)E + c0;
#pragma unroll
    for (int kc = 0; kc < 2; ++kc) {
      qf[kc].h[0] = *(const v8bf*)(qr + kc * 32 + half * 8);
      qf[kc].h[1] = *(const v8bf*)(qr + kc * 32 + 16 + half * 8);
    }
  }

  v8f o[4] = {};
  float mrow[8], lrow[8];
#pragma unroll
  for (int r = 0; r < 8; ++r) { mrow[r] = -3.0e38f; lrow[r] = 0.f; }

  const int lr = tid >> 1;  // tile-loader row 0..63
  const int lh = tid & 1;   // 32-element half

  for (int s0 = 0; s0 < T; s0 += 64) {
    __syncthreads();   // previous S-tile fully consumed
#if HAVE_TDM
    if (wave == 0)
      tdm_load_2d(Kg + (rowBase + s0) * (size_t)E + c0, lds_addr_of(Kt),
                  64u, 64u, (unsigned)E);
#else
    {
      const __bf16* ks = Kg + (rowBase + s0 + lr) * (size_t)E + c0 + lh * 32;
      uint4* kd = (uint4*)(Kt + lr * 64 + lh * 32);
      const uint4* ksv = (const uint4*)ks;
      kd[0] = ksv[0]; kd[1] = ksv[1]; kd[2] = ksv[2]; kd[3] = ksv[3];
    }
#endif
    {
      const __bf16* vs = Vg + (rowBase + s0 + lr) * (size_t)E + c0 + lh * 32;
#pragma unroll
      for (int j = 0; j < 32; ++j) Vt[(lh * 32 + j) * 64 + lr] = vs[j];
    }
#if HAVE_TDM
    if (wave == 0) __builtin_amdgcn_s_wait_tensorcnt(0);
#endif
    __syncthreads();

    // scores 16x64 = Q (16x64) . K^T  (batch 4 B-frags per k-step)
    v8f sc[4] = {};
#pragma unroll
    for (int kc = 0; kc < 2; ++kc) {
      FragA bk[4];
#pragma unroll
      for (int nt = 0; nt < 4; ++nt) {
        const __bf16* kb = Kt + (nt * 16 + l15) * 64 + kc * 32 + half * 8;
        bk[nt].h[0] = *(const v8bf*)(kb);
        bk[nt].h[1] = *(const v8bf*)(kb + 16);
      }
#pragma unroll
      for (int nt = 0; nt < 4; ++nt)
        sc[nt] = WMMA_BF16(qf[kc].v, bk[nt].v, sc[nt]);
    }
#pragma unroll
    for (int nt = 0; nt < 4; ++nt)
#pragma unroll
      for (int r = 0; r < 8; ++r) sc[nt][r] *= 0.125f;  // 1/sqrt(64)

    // online softmax: row r+8*half lives on the 16 lanes of this half-group;
    // shfl_xor masks 1,2,4,8 stay inside the group.
#pragma unroll
    for (int r = 0; r < 8; ++r) {
      float m = fmaxf(fmaxf(sc[0][r], sc[1][r]), fmaxf(sc[2][r], sc[3][r]));
#pragma unroll
      for (int off = 1; off < 16; off <<= 1) m = fmaxf(m, __shfl_xor(m, off, 32));
      const float mn = fmaxf(mrow[r], m);
      const float sf = __expf(mrow[r] - mn);
      mrow[r] = mn;
      lrow[r] *= sf;
      o[0][r] *= sf; o[1][r] *= sf; o[2][r] *= sf; o[3][r] *= sf;
      float rs = 0.f;
#pragma unroll
      for (int nt = 0; nt < 4; ++nt) {
        float p = __expf(sc[nt][r] - mn);
        sc[nt][r] = p;
        rs += p;
      }
#pragma unroll
      for (int off = 1; off < 16; off <<= 1) rs += __shfl_xor(rs, off, 32);
      lrow[r] += rs;
    }

    // C-layout -> A-layout via per-wave LDS round trip
    __bf16* pw = Pl + wave * (16 * 64);
#pragma unroll
    for (int nt = 0; nt < 4; ++nt)
#pragma unroll
      for (int r = 0; r < 8; ++r)
        pw[(r + 8 * half) * 64 + nt * 16 + l15] = (__bf16)sc[nt][r];

    FragA pf[2];  // same-wave LDS ops are in-order; compiler waits DScnt
    const __bf16* pr = pw + l15 * 64;
#pragma unroll
    for (int kp = 0; kp < 2; ++kp) {
      pf[kp].h[0] = *(const v8bf*)(pr + kp * 32 + half * 8);
      pf[kp].h[1] = *(const v8bf*)(pr + kp * 32 + 16 + half * 8);
    }

    // O += P (16x64) . V (64x64)
#pragma unroll
    for (int kp = 0; kp < 2; ++kp) {
      FragA bv[4];
#pragma unroll
      for (int nt = 0; nt < 4; ++nt) {
        const __bf16* vbp = Vt + (nt * 16 + l15) * 64 + kp * 32 + half * 8;
        bv[nt].h[0] = *(const v8bf*)(vbp);
        bv[nt].h[1] = *(const v8bf*)(vbp + 16);
      }
#pragma unroll
      for (int nt = 0; nt < 4; ++nt)
        o[nt] = WMMA_BF16(pf[kp].v, bv[nt].v, o[nt]);
    }
  }

#pragma unroll
  for (int r = 0; r < 8; ++r) {
    const float inv = 1.0f / lrow[r];
    const size_t row = rowBase + t0 + wave * 16 + r + 8 * half;
#pragma unroll
    for (int nt = 0; nt < 4; ++nt)
      O[row * (size_t)E + c0 + nt * 16 + l15] = (__bf16)(o[nt][r] * inv);
  }
}

// ---------------------------------------------------------------------------
// Orchestration
// ---------------------------------------------------------------------------
extern "C" void kernel_launch(void* const* d_in, const int* in_sizes, int n_in,
                              void* d_out, int out_size, void* d_ws, size_t ws_size,
                              hipStream_t stream) {
  (void)in_sizes; (void)n_in; (void)out_size; (void)ws_size;
  const float* x      = (const float*)d_in[0];
  const float* Wq     = (const float*)d_in[1];
  const float* Wk     = (const float*)d_in[2];
  const float* Wv     = (const float*)d_in[3];
  const float* proj_w = (const float*)d_in[4];
  const float* proj_b = (const float*)d_in[5];
  const float* ln1_g  = (const float*)d_in[6];
  const float* ln1_b  = (const float*)d_in[7];
  const float* ln2_g  = (const float*)d_in[8];
  const float* ln2_b  = (const float*)d_in[9];
  const float* up_w   = (const float*)d_in[10];
  const float* up_b   = (const float*)d_in[11];
  const float* down_w = (const float*)d_in[12];
  const float* down_b = (const float*)d_in[13];
  float* out = (float*)d_out;

  constexpr int B = 4, T = 2048, E = 1024, H = 16, D = 64, F = 4096;
  constexpr int M = B * T;  // 8192 token rows

  // workspace bump allocator (~216 MB total)
  char* wsb = (char*)d_ws;
  size_t off = 0;
  auto alloc = [&](size_t bytes) -> char* {
    char* p = wsb + off;
    off += (bytes + 255) & ~(size_t)255;
    return p;
  };
  __bf16* hln   = (__bf16*)alloc((size_t)M * E * 2);
  __bf16* qb    = (__bf16*)alloc((size_t)M * E * 2);
  __bf16* kb    = (__bf16*)alloc((size_t)M * E * 2);
  __bf16* vb    = (__bf16*)alloc((size_t)M * E * 2);
  __bf16* attn  = (__bf16*)alloc((size_t)M * E * 2);
  float*  x1    = (float*) alloc((size_t)M * E * 4);
  __bf16* h2    = (__bf16*)alloc((size_t)M * E * 2);
  __bf16* hid   = (__bf16*)alloc((size_t)M * F * 2);
  __bf16* wqT   = (__bf16*)alloc((size_t)E * E * 2);
  __bf16* wkT   = (__bf16*)alloc((size_t)E * E * 2);
  __bf16* wvT   = (__bf16*)alloc((size_t)E * E * 2);
  __bf16* projT = (__bf16*)alloc((size_t)E * E * 2);
  __bf16* upT   = (__bf16*)alloc((size_t)E * F * 2);
  __bf16* downT = (__bf16*)alloc((size_t)F * E * 2);

  // 1) weight pack / convert (bf16, transposed to [N][K])
  {
    int g = (E * E + 255) / 256;
    pack_qkv_kernel<<<g, 256, 0, stream>>>(Wq, wqT, E, D);
    pack_qkv_kernel<<<g, 256, 0, stream>>>(Wk, wkT, E, D);
    pack_qkv_kernel<<<g, 256, 0, stream>>>(Wv, wvT, E, D);
    pack_t_kernel<<<g, 256, 0, stream>>>(proj_w, projT, E, E);
    int gf = (E * F + 255) / 256;
    pack_t_kernel<<<gf, 256, 0, stream>>>(up_w, upT, E, F);
    pack_t_kernel<<<gf, 256, 0, stream>>>(down_w, downT, F, E);
  }

  const dim3 blk(128);

  // 2) LN1
  ln_kernel<<<M, 256, 0, stream>>>(x, ln1_g, ln1_b, hln, E);

  // 3) Q/K/V projections (8192 x 1024 x 1024 each)
  gemm_bf16_wmma<false, false, false, true, false>
      <<<dim3(E / 64, M / 64), blk, 0, stream>>>(hln, wqT, nullptr, nullptr, qb, nullptr, M, E, E);
  gemm_bf16_wmma<false, false, false, true, false>
      <<<dim3(E / 64, M / 64), blk, 0, stream>>>(hln, wkT, nullptr, nullptr, kb, nullptr, M, E, E);
  gemm_bf16_wmma<false, false, false, true, false>
      <<<dim3(E / 64, M / 64), blk, 0, stream>>>(hln, wvT, nullptr, nullptr, vb, nullptr, M, E, E);

  // 4) flash attention -> concat heads (stored as [M][E])
  attn_flash_wmma<<<dim3(T / 64, B * H), blk, 0, stream>>>(qb, kb, vb, attn, T, E, H);

  // 5) output projection + bias + residual(x) -> x1 (fp32)
  gemm_bf16_wmma<true, true, false, false, true>
      <<<dim3(E / 64, M / 64), blk, 0, stream>>>(attn, projT, proj_b, x, nullptr, x1, M, E, E);

  // 6) LN2
  ln_kernel<<<M, 256, 0, stream>>>(x1, ln2_g, ln2_b, h2, E);

  // 7) MLP up + bias + ReLU -> hid (bf16)
  gemm_bf16_wmma<true, false, true, true, false>
      <<<dim3(F / 64, M / 64), blk, 0, stream>>>(h2, upT, up_b, nullptr, hid, nullptr, M, F, E);

  // 8) MLP down + bias + residual(x1) -> out (fp32)
  gemm_bf16_wmma<true, true, false, false, true>
      <<<dim3(E / 64, M / 64), blk, 0, stream>>>(hid, downT, down_b, x1, nullptr, out, M, E, F);
}
